// Cross_Att_34840774705676
// MI455X (gfx1250) — matrix operand order repaired and checked by reading, not compiled, not run on hardware
//
#include <hip/hip_runtime.h>

#ifndef __has_builtin
#define __has_builtin(x) 0
#endif

#if __has_builtin(__builtin_amdgcn_global_load_async_to_lds_b128)
#define HAVE_ASYNC_LDS 1
#else
#define HAVE_ASYNC_LDS 0
#endif

// ---------------------------------------------------------------------------
// Types for CDNA5 WMMA (wave32): bf16 A/B fragments (16 bf16 = 8 VGPRs/lane),
// f32 C/D (8 VGPRs/lane).
// ---------------------------------------------------------------------------
typedef __attribute__((ext_vector_type(16))) __bf16 v16bf;
typedef __attribute__((ext_vector_type(2)))  __bf16 bf16x2;
typedef __attribute__((ext_vector_type(8)))  float  v8f;
typedef __attribute__((ext_vector_type(4)))  int    i4v;

union FragAB {
    v16bf v;
    uint4 q[2];
};

// Native bf16 conversion (v_cvt_pk_bf16_f32 confirmed in disasm).
static __device__ __forceinline__ unsigned int cvt2_bf16(float a, float b) {
    bf16x2 h;
    h[0] = (__bf16)a;
    h[1] = (__bf16)b;
    return __builtin_bit_cast(unsigned int, h);
}
static __device__ __forceinline__ unsigned short cvt1_bf16(float a) {
    return __builtin_bit_cast(unsigned short, (__bf16)a);
}

#define IN_DIM  1024
#define NK_DIM  512      // dim_k == dim_v
#define NROWS   8192
#define NTILES  (NROWS / 32)

#if HAVE_ASYNC_LDS
// Async global -> LDS 16-byte copy (ASYNCcnt-tracked, bypasses VGPRs).
static __device__ __forceinline__ void async_cp16(const void* g, void* l) {
    __builtin_amdgcn_global_load_async_to_lds_b128(
        (__attribute__((address_space(1))) i4v*)g,
        (__attribute__((address_space(3))) i4v*)l,
        0, 0);
}
// Threshold wait on ASYNCcnt; argument must be an integer constant expression.
template <int N>
static __device__ __forceinline__ void wait_async() {
#if __has_builtin(__builtin_amdgcn_s_wait_asynccnt)
    __builtin_amdgcn_s_wait_asynccnt(N);
#else
    asm volatile("s_wait_asynccnt %0" :: "i"(N) : "memory");
#endif
}
#endif

// ---------------------------------------------------------------------------
// Projection: out(bf16) = X[rows,1024] @ W[512,1024]^T + b
// Grid: (rows/64, 512/64), block = 128 (4 waves). Each wave -> 16 rows x 64 cols.
// Compile-time output strides so V can be stored transposed with cheap
// address math (no v_mul_u64 chains in the epilogue).
// ---------------------------------------------------------------------------
template <int SM, int SN>
__global__ __launch_bounds__(128)
void proj_kernel(const float* __restrict__ X, const float* __restrict__ W,
                 const float* __restrict__ bias, unsigned short* __restrict__ out) {
    __shared__ __align__(16) unsigned short As[64 * 32];
    __shared__ __align__(16) unsigned short Bs[64 * 32];

    const int tid  = threadIdx.x;
    const int lane = tid & 31;
    const int wave = tid >> 5;
    const int m0   = blockIdx.x * 64;
    const int n0   = blockIdx.y * 64;

    v8f acc[4];
#pragma unroll
    for (int t = 0; t < 4; ++t)
#pragma unroll
        for (int i = 0; i < 8; ++i) acc[t][i] = 0.f;

    const int lr = tid >> 1;   // row 0..63 within tile
    const int lh = tid & 1;    // which 16-float half of the 32-wide K chunk
    const int fm   = lane & 15;
    const int half = lane >> 4;

    for (int k0 = 0; k0 < IN_DIM; k0 += 32) {
        __syncthreads();
        {   // stage 16 f32 -> 16 bf16 per thread, packed b128 stores
            const float4* srcA =
                (const float4*)(X + (size_t)(m0 + lr) * IN_DIM + k0 + lh * 16);
            float4 a0 = srcA[0], a1 = srcA[1], a2 = srcA[2], a3 = srcA[3];
            uint4 o0, o1;
            o0.x = cvt2_bf16(a0.x, a0.y); o0.y = cvt2_bf16(a0.z, a0.w);
            o0.z = cvt2_bf16(a1.x, a1.y); o0.w = cvt2_bf16(a1.z, a1.w);
            o1.x = cvt2_bf16(a2.x, a2.y); o1.y = cvt2_bf16(a2.z, a2.w);
            o1.z = cvt2_bf16(a3.x, a3.y); o1.w = cvt2_bf16(a3.z, a3.w);
            uint4* dstA = (uint4*)&As[lr * 32 + lh * 16];
            dstA[0] = o0; dstA[1] = o1;

            const float4* srcB =
                (const float4*)(W + (size_t)(n0 + lr) * IN_DIM + k0 + lh * 16);
            float4 b0 = srcB[0], b1 = srcB[1], b2 = srcB[2], b3 = srcB[3];
            uint4 p0, p1;
            p0.x = cvt2_bf16(b0.x, b0.y); p0.y = cvt2_bf16(b0.z, b0.w);
            p0.z = cvt2_bf16(b1.x, b1.y); p0.w = cvt2_bf16(b1.z, b1.w);
            p1.x = cvt2_bf16(b2.x, b2.y); p1.y = cvt2_bf16(b2.z, b2.w);
            p1.z = cvt2_bf16(b3.x, b3.y); p1.w = cvt2_bf16(b3.z, b3.w);
            uint4* dstB = (uint4*)&Bs[lr * 32 + lh * 16];
            dstB[0] = p0; dstB[1] = p1;
        }
        __syncthreads();

        FragAB a;
        {   // A: rows wave*16+fm, K chunks [half*8,+8) and [16+half*8,+8)
            const unsigned short* p = &As[(wave * 16 + fm) * 32];
            a.q[0] = *(const uint4*)(p + half * 8);
            a.q[1] = *(const uint4*)(p + 16 + half * 8);
        }
#pragma unroll
        for (int t = 0; t < 4; ++t) {
            FragAB b;   // B: col n = t*16+fm, K chunks [half*8,+8) and [16+half*8,+8)
            const unsigned short* p = &Bs[(t * 16 + fm) * 32];
            b.q[0] = *(const uint4*)(p + half * 8);
            b.q[1] = *(const uint4*)(p + 16 + half * 8);
            acc[t] = __builtin_amdgcn_wmma_f32_16x16x32_bf16(
                false, a.v, false, b.v, (short)0, acc[t], false, false);
        }
    }

    // bias + bf16 store (compile-time strides; V goes out transposed)
#pragma unroll
    for (int t = 0; t < 4; ++t) {
        const int ncol = n0 + t * 16 + fm;
        const float bv = bias[ncol];
#pragma unroll
        for (int r = 0; r < 8; ++r) {
            const int mrow = m0 + wave * 16 + half * 8 + r;
            out[(size_t)mrow * SM + (size_t)ncol * SN] = cvt1_bf16(acc[t][r] + bv);
        }
    }
}

// ---------------------------------------------------------------------------
// Flash attention: out = softmax(Q K^T) * (1/sqrt(512)) @ V
// Grid: 8192/64 blocks, 128 threads (4 waves). Wave w owns query rows
// q0+w*16..+16, full dv=512 accumulator (32 C-tiles). Keys tiled by 32,
// double-buffered in LDS with async-to-LDS prefetch one tile ahead.
// ---------------------------------------------------------------------------
__global__ __launch_bounds__(128)
void attn_kernel(const unsigned short* __restrict__ Qb,
                 const unsigned short* __restrict__ Kb,
                 const unsigned short* __restrict__ Vt,   // [dv][key] transposed
                 float* __restrict__ out) {
    __shared__ __align__(16) unsigned short Qs[64 * 512];       // 64 KB
    __shared__ __align__(16) unsigned short Ks[2 * 32 * 512];   // 2x32 KB
    __shared__ __align__(16) unsigned short Vs[2 * 512 * 32];   // 2x32 KB (dv-major)
    __shared__ __align__(16) unsigned short Ps[4][16 * 32];     // per-wave P tile

    const int tid  = threadIdx.x;
    const int lane = tid & 31;
    const int wave = tid >> 5;
    const int q0   = blockIdx.x * 64;
    const int fm   = lane & 15;
    const int half = lane >> 4;

    const uint4* Kb4 = (const uint4*)Kb;
    const uint4* Vt4 = (const uint4*)Vt;

    // Stage tile -> LDS buffer (16 K-chunks + 16 V-chunks per thread).
    auto stage = [&](int tile, int buf) {
        uint4* kd = (uint4*)(Ks + buf * (32 * 512));
        uint4* vd = (uint4*)(Vs + buf * (512 * 32));
        const int kt = tile * 32;
#pragma unroll
        for (int i = 0; i < 16; ++i) {
            const int j  = tid + i * 128;          // 0..2047
            const int dv = j >> 2, c = j & 3;
#if HAVE_ASYNC_LDS
            async_cp16(&Kb4[(size_t)kt * 64 + j], &kd[j]);
            async_cp16(&Vt4[(size_t)dv * (NROWS / 8) + (kt >> 3) + c], &vd[j]);
#else
            kd[j] = Kb4[(size_t)kt * 64 + j];
            vd[j] = Vt4[(size_t)dv * (NROWS / 8) + (kt >> 3) + c];
#endif
        }
    };

    // Stage this block's Q rows once (64x512 bf16)
    {
        const uint4* src = (const uint4*)(Qb + (size_t)q0 * 512);
        uint4* dst = (uint4*)Qs;
#pragma unroll
        for (int i = 0; i < 32; ++i) {
#if HAVE_ASYNC_LDS
            async_cp16(&src[tid + i * 128], &dst[tid + i * 128]);
#else
            dst[tid + i * 128] = src[tid + i * 128];
#endif
        }
    }
    stage(0, 0);   // prime buffer 0

    v8f acc[32];
#pragma unroll
    for (int t = 0; t < 32; ++t)
#pragma unroll
        for (int i = 0; i < 8; ++i) acc[t][i] = 0.f;

    float rmax[8], rsum[8];
#pragma unroll
    for (int r = 0; r < 8; ++r) { rmax[r] = -3.0e38f; rsum[r] = 0.f; }

    for (int it = 0; it < NTILES; ++it) {
        const int cur = it & 1;
        const bool haveNext = (it + 1) < NTILES;
        if (haveNext) stage(it + 1, cur ^ 1);      // prefetch next tile
#if HAVE_ASYNC_LDS
        // Async ops retire in order: <=32 outstanding means the current
        // tile's 32 copies per thread (and Q on iter 0) have landed.
        if (haveNext) wait_async<32>(); else wait_async<0>();
#endif
        __syncthreads();   // current tile visible to all waves

        const unsigned short* KsC = Ks + cur * (32 * 512);
        const unsigned short* VsC = Vs + cur * (512 * 32);

        // ---- scores: S[16 q][32 keys] = Q_tile @ K_tile^T ----
        v8f s0, s1;
#pragma unroll
        for (int i = 0; i < 8; ++i) { s0[i] = 0.f; s1[i] = 0.f; }
#pragma unroll
        for (int dk = 0; dk < 512; dk += 32) {
            FragAB a;
            const unsigned short* qp = &Qs[(wave * 16 + fm) * 512 + dk];
            a.q[0] = *(const uint4*)(qp + half * 8);
            a.q[1] = *(const uint4*)(qp + 16 + half * 8);
            FragAB b0, b1;
            const unsigned short* kp0 = &KsC[fm * 512 + dk];
            b0.q[0] = *(const uint4*)(kp0 + half * 8);
            b0.q[1] = *(const uint4*)(kp0 + 16 + half * 8);
            const unsigned short* kp1 = &KsC[(16 + fm) * 512 + dk];
            b1.q[0] = *(const uint4*)(kp1 + half * 8);
            b1.q[1] = *(const uint4*)(kp1 + 16 + half * 8);
            s0 = __builtin_amdgcn_wmma_f32_16x16x32_bf16(
                false, a.v, false, b0.v, (short)0, s0, false, false);
            s1 = __builtin_amdgcn_wmma_f32_16x16x32_bf16(
                false, a.v, false, b1.v, (short)0, s1, false, false);
        }

        // ---- online softmax (rows m = half*8 + r, cols across 16 lanes) ----
        float tmax[8];
#pragma unroll
        for (int r = 0; r < 8; ++r) tmax[r] = fmaxf(s0[r], s1[r]);
#pragma unroll
        for (int mm = 1; mm <= 8; mm <<= 1)
#pragma unroll
            for (int r = 0; r < 8; ++r)
                tmax[r] = fmaxf(tmax[r], __shfl_xor(tmax[r], mm, 32));

        float scl[8];
#pragma unroll
        for (int r = 0; r < 8; ++r) {
            const float mnew = fmaxf(rmax[r], tmax[r]);
            scl[r] = __expf(rmax[r] - mnew);
            rmax[r] = mnew;
        }

        float ts[8];
#pragma unroll
        for (int r = 0; r < 8; ++r) {
            const float p0 = __expf(s0[r] - rmax[r]);
            const float p1 = __expf(s1[r] - rmax[r]);
            ts[r] = p0 + p1;
            const int m = half * 8 + r;
            Ps[wave][m * 32 + fm]      = cvt1_bf16(p0);
            Ps[wave][m * 32 + 16 + fm] = cvt1_bf16(p1);
        }
#pragma unroll
        for (int mm = 1; mm <= 8; mm <<= 1)
#pragma unroll
            for (int r = 0; r < 8; ++r) ts[r] += __shfl_xor(ts[r], mm, 32);
#pragma unroll
        for (int r = 0; r < 8; ++r) rsum[r] = rsum[r] * scl[r] + ts[r];

        // rescale running accumulator
#pragma unroll
        for (int t = 0; t < 32; ++t)
#pragma unroll
            for (int r = 0; r < 8; ++r) acc[t][r] *= scl[r];

        // wave-coherent LDS ordering for the P bounce (no block barrier needed)
        asm volatile("s_wait_dscnt 0" ::: "memory");

        // ---- P @ V_tile: accumulate into all 32 dv tiles ----
        FragAB pf;
        {
            const unsigned short* pp = &Ps[wave][fm * 32];
            pf.q[0] = *(const uint4*)(pp + half * 8);
            pf.q[1] = *(const uint4*)(pp + 16 + half * 8);
        }
#pragma unroll
        for (int t = 0; t < 32; ++t) {
            FragAB vb;
            const unsigned short* vp = &VsC[(t * 16 + fm) * 32];
            vb.q[0] = *(const uint4*)(vp + half * 8);
            vb.q[1] = *(const uint4*)(vp + 16 + half * 8);
            acc[t] = __builtin_amdgcn_wmma_f32_16x16x32_bf16(
                false, pf.v, false, vb.v, (short)0, acc[t], false, false);
        }
        __syncthreads();   // all waves done with this buffer before reuse
    }

    // ---- epilogue: out = acc / rowsum * (1/sqrt(512)), f32 ----
    const float norm = 0.044194173824159216f;  // 1/sqrt(512)
#pragma unroll
    for (int r = 0; r < 8; ++r) {
        const float inv = norm / rsum[r];
        const int mrow = q0 + wave * 16 + half * 8 + r;
#pragma unroll
        for (int t = 0; t < 32; ++t)
            out[(size_t)mrow * NK_DIM + t * 16 + fm] = acc[t][r] * inv;
    }
}

// ---------------------------------------------------------------------------
extern "C" void kernel_launch(void* const* d_in, const int* in_sizes, int n_in,
                              void* d_out, int out_size, void* d_ws, size_t ws_size,
                              hipStream_t stream) {
    (void)in_sizes; (void)n_in; (void)out_size; (void)ws_size;
    const float* q  = (const float*)d_in[0];
    const float* k  = (const float*)d_in[1];
    const float* v  = (const float*)d_in[2];
    const float* Wq = (const float*)d_in[3];
    const float* bq = (const float*)d_in[4];
    const float* Wk = (const float*)d_in[5];
    const float* bk = (const float*)d_in[6];
    const float* Wv = (const float*)d_in[7];
    const float* bv = (const float*)d_in[8];

    unsigned short* Qb = (unsigned short*)d_ws;                    // [8192][512] bf16
    unsigned short* Kb = Qb + (size_t)NROWS * NK_DIM;              // [8192][512] bf16
    unsigned short* Vt = Kb + (size_t)NROWS * NK_DIM;              // [512][8192] bf16 (transposed)

    dim3 gproj(NROWS / 64, NK_DIM / 64);
    proj_kernel<NK_DIM, 1><<<gproj, 128, 0, stream>>>(q, Wq, bq, Qb);
    proj_kernel<NK_DIM, 1><<<gproj, 128, 0, stream>>>(k, Wk, bk, Kb);
    proj_kernel<1, NROWS><<<gproj, 128, 0, stream>>>(v, Wv, bv, Vt);

    attn_kernel<<<dim3(NROWS / 64), 128, 0, stream>>>(Qb, Kb, Vt, (float*)d_out);
}
